// SingleHeadAttention_44976897524631
// MI455X (gfx1250) — matrix operand order repaired and checked by reading, not compiled, not run on hardware
//
#include <hip/hip_runtime.h>
#include <stdint.h>

// Problem constants (from reference): B=16, S=2048, D=768, HS=64
#define B_  16
#define S_  2048
#define D_  768
#define HS_ 64

typedef __attribute__((ext_vector_type(16))) __bf16 v16bf;
typedef __attribute__((ext_vector_type(8)))  float  v8f;

union BfFrag  { v16bf v; uint32_t u[8]; };
union AccFrag { v8f   v; float    f[8]; };

__device__ __forceinline__ uint16_t f2bf(float f) {
  uint32_t u = __builtin_bit_cast(uint32_t, f);
  u += 0x7FFFu + ((u >> 16) & 1u);        // round-to-nearest-even
  return (uint16_t)(u >> 16);
}
__device__ __forceinline__ uint32_t pack2bf(float f0, float f1) {
  return (uint32_t)f2bf(f0) | ((uint32_t)f2bf(f1) << 16);
}

// A-matrix (16-bit, 16x32) per-lane K offset for VGPR r (ISA 7.12.2):
// lanes 0-15: v0..3 -> K={0..7}, v4..7 -> K={16..23}; lanes 16-31: +8
__device__ __forceinline__ int a_koff(int r, bool hi) {
  return ((r & 4) ? 16 : 0) + (hi ? 8 : 0) + (r & 3) * 2;
}
// B-matrix (16-bit, 32x16): same K order as A: v0..3 -> K={0..7}, v4..7 -> K={16..23}; lanes 16-31: +8
__device__ __forceinline__ int b_koff(int r, bool hi) {
  return ((r & 4) ? 16 : 0) + (hi ? 8 : 0) + (r & 3) * 2;
}

// ---------------------------------------------------------------------------
// Kernel 0: one-shot weight prep.  Wt[m][col][kk] = bf16( W_m[kk][col] )
// (transposed so GEMM B-fragments are single contiguous u32 loads).
// ---------------------------------------------------------------------------
__global__ __launch_bounds__(256)
void prep_w_kernel(const float* __restrict__ Wq, const float* __restrict__ Wk,
                   const float* __restrict__ Wv, uint16_t* __restrict__ Wt)
{
  int idx = blockIdx.x * 256 + threadIdx.x;          // 3 * 64 * 768 total
  if (idx >= 3 * HS_ * D_) return;
  int m   = idx / (HS_ * D_);
  int rem = idx % (HS_ * D_);
  int col = rem / D_;
  int kk  = rem % D_;
  const float* W = (m == 0) ? Wq : (m == 1) ? Wk : Wv;
  Wt[idx] = f2bf(W[(size_t)kk * HS_ + col]);
}

// ---------------------------------------------------------------------------
// Kernel 1: fused QKV projection — reads x ONCE, produces all three outputs.
// Q,K row-major bf16 [B*S][64];  V transposed bf16 vT[b][64][2048].
// grid = 512, block = 128 (4 waves); wave tile = 16 rows x 64 cols x 3 mats.
// ---------------------------------------------------------------------------
__global__ __launch_bounds__(128)
void qkv_fused_kernel(const float* __restrict__ x,
                      const uint16_t* __restrict__ Wt,
                      const float* __restrict__ bq, const float* __restrict__ bk,
                      const float* __restrict__ bv,
                      uint16_t* __restrict__ qo, uint16_t* __restrict__ ko,
                      uint16_t* __restrict__ vT)
{
  const int wave = threadIdx.x >> 5;
  const int lane = threadIdx.x & 31;
  const int lo   = lane & 15;
  const bool hi  = lane >= 16;
  const int m0   = (blockIdx.x * 4 + wave) * 16;     // row-tile base [0,32768)

  AccFrag acc[3][4];
#pragma unroll
  for (int m = 0; m < 3; ++m)
#pragma unroll
    for (int t = 0; t < 4; ++t)
#pragma unroll
      for (int r = 0; r < 8; ++r) acc[m][t].f[r] = 0.0f;

  const float* xrow = x + (size_t)(m0 + lo) * D_;

  for (int kb = 0; kb < D_; kb += 32) {
    // A fragment: 16x32 tile of x, converted to bf16 on the fly
    BfFrag a;
#pragma unroll
    for (int r = 0; r < 8; ++r) {
      int koff = a_koff(r, hi);
      a.u[r] = pack2bf(xrow[kb + koff], xrow[kb + koff + 1]);
    }
#pragma unroll
    for (int m = 0; m < 3; ++m) {
      const uint16_t* wm = Wt + (size_t)m * HS_ * D_;
#pragma unroll
      for (int t = 0; t < 4; ++t) {
        BfFrag bfr;                                  // B: 32x16 of W (pre-bf16)
        int col = t * 16 + lo;
#pragma unroll
        for (int r = 0; r < 8; ++r)
          bfr.u[r] = *(const uint32_t*)(wm + (size_t)col * D_ + kb + b_koff(r, hi));
        acc[m][t].v = __builtin_amdgcn_wmma_f32_16x16x32_bf16(
            false, a.v, false, bfr.v, (short)0, acc[m][t].v, false, false);
      }
    }
  }

  // ---- stores.  D layout: vgpr r -> row = r + (hi?8:0), col = lane&15 -----
#pragma unroll
  for (int t = 0; t < 4; ++t) {
    int col = t * 16 + lo;
    float bcq = bq[col], bck = bk[col];
#pragma unroll
    for (int r = 0; r < 8; ++r) {
      size_t row = (size_t)(m0 + r + (hi ? 8 : 0));
      qo[row * HS_ + col] = f2bf(acc[0][t].f[r] + bcq);
      ko[row * HS_ + col] = f2bf(acc[1][t].f[r] + bck);
    }
  }
  // V transposed: vT[b][col][s]; lane owns 8 consecutive s -> 4 u32 stores
  {
    const int b  = m0 >> 11;                 // /2048
    const int s0 = (m0 & (S_ - 1)) + (hi ? 8 : 0);
#pragma unroll
    for (int t = 0; t < 4; ++t) {
      int col = t * 16 + lo;
      float bcv = bv[col];
      uint16_t* dst = vT + ((size_t)b * HS_ + col) * S_ + s0;
#pragma unroll
      for (int r = 0; r < 4; ++r) {
        uint32_t p = pack2bf(acc[2][t].f[2 * r] + bcv, acc[2][t].f[2 * r + 1] + bcv);
        *(uint32_t*)(dst + 2 * r) = p;
      }
    }
  }
}

// ---------------------------------------------------------------------------
// Kernel 2: causal flash attention.  Each wave owns one 16-row query tile;
// streams 32-key blocks with online softmax; P transposed via 1KB LDS/wave.
// V is consumed from the transposed vT layout (contiguous u32 fragments).
// grid = 512, block = 128 (4 waves).
// ---------------------------------------------------------------------------
__global__ __launch_bounds__(128)
void flash_attn_kernel(const uint16_t* __restrict__ q,
                       const uint16_t* __restrict__ k,
                       const uint16_t* __restrict__ vT,
                       float* __restrict__ out)
{
  __shared__ uint16_t Pld[4][16 * 32];   // per-wave P staging (bf16)

  const int wave = threadIdx.x >> 5;
  const int lane = threadIdx.x & 31;
  const int lo   = lane & 15;
  const bool hi  = lane >= 16;

  const int tile = blockIdx.x * 4 + wave;   // 0 .. 2047
  const int b    = tile >> 7;               // batch
  const int qb   = tile & 127;              // query block (16 rows each)
  const size_t base  = (size_t)b * S_ * HS_;    // q/k row-major base
  const size_t vbase = (size_t)b * HS_ * S_;    // vT base

  // Q A-fragments (16 rows x 64 K) -> two 16x32 fragments, loaded once.
  BfFrag qa[2];
  const uint16_t* qrow = q + base + (size_t)(qb * 16 + lo) * HS_;
#pragma unroll
  for (int h = 0; h < 2; ++h)
#pragma unroll
    for (int r = 0; r < 8; ++r) {
      int koff = h * 32 + a_koff(r, hi);
      qa[h].u[r] = *(const uint32_t*)(qrow + koff);   // 2 consecutive bf16
    }

  AccFrag oacc[4];
#pragma unroll
  for (int t = 0; t < 4; ++t)
#pragma unroll
    for (int r = 0; r < 8; ++r) oacc[t].f[r] = 0.0f;

  float mrow[8], lrow[8];
#pragma unroll
  for (int r = 0; r < 8; ++r) { mrow[r] = -__builtin_inff(); lrow[r] = 0.0f; }

  const float scale = 0.125f;               // 1/sqrt(64)
  const int jmax = (qb * 16 + 15) >> 5;     // last key block touching diagonal

  uint16_t* ps = &Pld[wave][0];

  for (int j = 0; j <= jmax; ++j) {
    if (j < jmax) {                          // prefetch next K/V block
      __builtin_prefetch(k + base + (size_t)((j + 1) * 32 + lo) * HS_, 0, 0);
      __builtin_prefetch(vT + vbase + (size_t)lo * S_ + (j + 1) * 32, 0, 0);
    }

    // ---- S = Q · K^T  (16 x 32), two 16x16 column tiles -------------------
    AccFrag s[2];
#pragma unroll
    for (int nt = 0; nt < 2; ++nt) {
#pragma unroll
      for (int r = 0; r < 8; ++r) s[nt].f[r] = 0.0f;
      const uint16_t* krow = k + base + (size_t)(j * 32 + nt * 16 + lo) * HS_;
#pragma unroll
      for (int h = 0; h < 2; ++h) {
        BfFrag kf;                           // B: K^T tile, 32x16
#pragma unroll
        for (int r = 0; r < 8; ++r) {
          int koff = h * 32 + b_koff(r, hi);
          kf.u[r] = *(const uint32_t*)(krow + koff);
        }
        s[nt].v = __builtin_amdgcn_wmma_f32_16x16x32_bf16(
            false, qa[h].v, false, kf.v, (short)0, s[nt].v, false, false);
      }
    }

    // ---- causal mask + scale, online softmax ------------------------------
    float c[8];
#pragma unroll
    for (int r = 0; r < 8; ++r) {
      int row = qb * 16 + r + (hi ? 8 : 0);
#pragma unroll
      for (int nt = 0; nt < 2; ++nt) {
        int col = j * 32 + nt * 16 + lo;
        s[nt].f[r] = (col <= row) ? s[nt].f[r] * scale : -__builtin_inff();
      }
      // row max across 32 columns (16 lanes per half x 2 tiles)
      float t = fmaxf(s[0].f[r], s[1].f[r]);
#pragma unroll
      for (int d = 1; d < 16; d <<= 1) t = fmaxf(t, __shfl_xor(t, d, 32));
      float mnew = fmaxf(mrow[r], t);
      c[r] = __expf(mrow[r] - mnew);         // rescale factor (0 on 1st iter)
      mrow[r] = mnew;

      float rsum = 0.0f;
#pragma unroll
      for (int nt = 0; nt < 2; ++nt) {
        float p = __expf(s[nt].f[r] - mnew); // exp(-inf) = 0 for masked cols
        s[nt].f[r] = p;
        rsum += p;
      }
#pragma unroll
      for (int d = 1; d < 16; d <<= 1) rsum += __shfl_xor(rsum, d, 32);
      lrow[r] = lrow[r] * c[r] + rsum;
    }

    // rescale running output accumulator
#pragma unroll
    for (int t = 0; t < 4; ++t)
#pragma unroll
      for (int r = 0; r < 8; ++r) oacc[t].f[r] *= c[r];

    // ---- transpose P (D layout -> A layout) through LDS -------------------
#pragma unroll
    for (int nt = 0; nt < 2; ++nt)
#pragma unroll
      for (int r = 0; r < 8; ++r) {
        int row = r + (hi ? 8 : 0);
        ps[row * 32 + nt * 16 + lo] = f2bf(s[nt].f[r]);
      }
    // same-wave DS ops are in order; compiler inserts s_wait_dscnt for loads
    BfFrag pa;
#pragma unroll
    for (int r = 0; r < 8; ++r) {
      int koff = a_koff(r, hi);
      pa.u[r] = *(const uint32_t*)(ps + lo * 32 + koff);
    }

    // ---- O += P · V  (K-dim = 32 keys), V from transposed layout ----------
#pragma unroll
    for (int t = 0; t < 4; ++t) {
      BfFrag vf;                             // B: 32x16 tile of V
      int col = t * 16 + lo;
      const uint16_t* vcol = vT + vbase + (size_t)col * S_ + j * 32;
#pragma unroll
      for (int r = 0; r < 8; ++r)
        vf.u[r] = *(const uint32_t*)(vcol + b_koff(r, hi)); // keys kk,kk+1
      oacc[t].v = __builtin_amdgcn_wmma_f32_16x16x32_bf16(
          false, pa.v, false, vf.v, (short)0, oacc[t].v, false, false);
    }
  }

  // ---- final normalize + store (fp32) -------------------------------------
#pragma unroll
  for (int t = 0; t < 4; ++t) {
    int col = t * 16 + lo;
#pragma unroll
    for (int r = 0; r < 8; ++r) {
      int row = qb * 16 + r + (hi ? 8 : 0);
      out[base + (size_t)row * HS_ + col] = oacc[t].f[r] / lrow[r];
    }
  }
}

// ---------------------------------------------------------------------------
extern "C" void kernel_launch(void* const* d_in, const int* in_sizes, int n_in,
                              void* d_out, int out_size, void* d_ws, size_t ws_size,
                              hipStream_t stream)
{
  const float* x  = (const float*)d_in[0];
  const float* Wq = (const float*)d_in[1];
  const float* bq = (const float*)d_in[2];
  const float* Wk = (const float*)d_in[3];
  const float* bk = (const float*)d_in[4];
  const float* Wv = (const float*)d_in[5];
  const float* bv = (const float*)d_in[6];

  // Workspace: bf16 q, k (row-major), vT (transposed), Wt (weights) ~12.3 MB
  const size_t elems = (size_t)B_ * S_ * HS_;
  uint16_t* q  = (uint16_t*)d_ws;
  uint16_t* k  = q + elems;
  uint16_t* vT = k + elems;
  uint16_t* Wt = vT + elems;

  // Weight prep: 3*64*768 elements, one thread each
  prep_w_kernel<<<(3 * HS_ * D_ + 255) / 256, 256, 0, stream>>>(Wq, Wk, Wv, Wt);

  // Fused QKV: 32768 rows / (16 rows/wave * 4 waves) = 512 blocks
  qkv_fused_kernel<<<512, 128, 0, stream>>>(x, Wt, bq, bk, bv, q, k, vT);

  // Flash attention: 16*128 query tiles / 4 waves per block = 512 blocks
  flash_attn_kernel<<<512, 128, 0, stream>>>(q, k, vT, (float*)d_out);
}
